// PointNetSetAbstractionMSG_75436805587149
// MI455X (gfx1250) — compile-verified, matrix operand order checked by reading, not run on hardware
//
#include <hip/hip_runtime.h>

// ---------------------------------------------------------------------------
// PointNet++ Set Abstraction (MSG) for MI455X / gfx1250
// B=4, N=4096, S=1024, C_IN=64; radii (.1,.2,.4), K=(16,32,128)
// MLPs: (64,64,128) (128,128,256) (128,128,256); BN(train) + ReLU; max over K
// GEMMs: v_wmma_f32_16x16x32_bf16, W staged in LDS, A fragments cached in
// registers so A is read from HBM exactly once per layer (min traffic).
// ---------------------------------------------------------------------------

typedef __attribute__((ext_vector_type(16))) __bf16 v16bf;
typedef __attribute__((ext_vector_type(8)))  __bf16 v8bf;
typedef __attribute__((ext_vector_type(8)))  float  v8f;

#define BB 4
#define NN 4096
#define SS 1024
#define CIN 64
#define BN_EPS 1e-5f

// ---------------- Farthest point sampling: one block per batch -------------
__global__ void fps_kernel(const float* __restrict__ xyz,
                           int* __restrict__ fps_idx,
                           float* __restrict__ new_xyz) {
  __shared__ float s_dist[NN];
  __shared__ float s_val[256];
  __shared__ int   s_idx[256];
  __shared__ int   s_far;
  const int b = blockIdx.x;
  const int t = threadIdx.x;
  const float* xb = xyz + (size_t)b * NN * 3;
  for (int n = t; n < NN; n += 256) s_dist[n] = 1e10f;
  if (t == 0) s_far = 0;
  __syncthreads();
  for (int it = 0; it < SS; ++it) {
    const int far = s_far;
    const float cx = xb[far * 3 + 0];
    const float cy = xb[far * 3 + 1];
    const float cz = xb[far * 3 + 2];
    if (t == 0) {
      fps_idx[b * SS + it] = far;
      new_xyz[((size_t)b * SS + it) * 3 + 0] = cx;
      new_xyz[((size_t)b * SS + it) * 3 + 1] = cy;
      new_xyz[((size_t)b * SS + it) * 3 + 2] = cz;
    }
    float bv = -1.0f; int bi = 0;
    for (int n = t; n < NN; n += 256) {
      const float dx = xb[n * 3 + 0] - cx;
      const float dy = xb[n * 3 + 1] - cy;
      const float dz = xb[n * 3 + 2] - cz;
      const float d  = dx * dx + dy * dy + dz * dz;
      const float nd = fminf(s_dist[n], d);
      s_dist[n] = nd;
      if (nd > bv) { bv = nd; bi = n; }
    }
    s_val[t] = bv; s_idx[t] = bi;
    __syncthreads();
    for (int off = 128; off > 0; off >>= 1) {
      if (t < off && s_val[t + off] > s_val[t]) {
        s_val[t] = s_val[t + off]; s_idx[t] = s_idx[t + off];
      }
      __syncthreads();
    }
    if (t == 0) s_far = s_idx[0];
    __syncthreads();
  }
}

// ---------------- Ball query (K nearest, outside-radius -> nearest) --------
__global__ void ball_query_kernel(const float* __restrict__ xyz,
                                  const float* __restrict__ new_xyz,
                                  int* __restrict__ idxbuf,
                                  float r2, int K) {
  __shared__ float s_dist[NN];
  __shared__ float s_val[256];
  __shared__ int   s_idx[256];
  __shared__ int   s_near;
  const int row = blockIdx.x;           // b*SS + s
  const int b = row / SS;
  const int t = threadIdx.x;
  const float cx = new_xyz[(size_t)row * 3 + 0];
  const float cy = new_xyz[(size_t)row * 3 + 1];
  const float cz = new_xyz[(size_t)row * 3 + 2];
  const float* xb = xyz + (size_t)b * NN * 3;
  for (int n = t; n < NN; n += 256) {
    const float dx = xb[n * 3 + 0] - cx;
    const float dy = xb[n * 3 + 1] - cy;
    const float dz = xb[n * 3 + 2] - cz;
    s_dist[n] = dx * dx + dy * dy + dz * dz;
  }
  __syncthreads();
  for (int k = 0; k < K; ++k) {
    float bv = 3.0e38f; int bi = 0;
    for (int n = t; n < NN; n += 256) {
      const float d = s_dist[n];
      if (d < bv) { bv = d; bi = n; }
    }
    s_val[t] = bv; s_idx[t] = bi;
    __syncthreads();
    for (int off = 128; off > 0; off >>= 1) {
      if (t < off && s_val[t + off] < s_val[t]) {
        s_val[t] = s_val[t + off]; s_idx[t] = s_idx[t + off];
      }
      __syncthreads();
    }
    if (t == 0) {
      const int   mi = s_idx[0];
      const float md = s_val[0];
      if (k == 0) s_near = mi;
      idxbuf[(size_t)row * K + k] = (md > r2) ? s_near : mi;
      s_dist[mi] = 3.0e38f;
    }
    __syncthreads();
  }
}

// ---------------- Gather + concat(rel_xyz, feats) -> bf16 A[M, Cpad] -------
__global__ void gather_kernel(const float* __restrict__ xyz,
                              const float* __restrict__ points,
                              const float* __restrict__ new_xyz,
                              const int* __restrict__ idxbuf,
                              __bf16* __restrict__ A,
                              int K, int Cpad, size_t total) {
  const size_t stride = (size_t)gridDim.x * 256u;
  for (size_t tgl = (size_t)blockIdx.x * 256u + threadIdx.x; tgl < total; tgl += stride) {
    const int    c  = (int)(tgl % Cpad);
    const size_t m  = tgl / Cpad;       // b*SS*K + s*K + k
    const size_t bs = m / K;            // b*SS + s
    const int    b  = (int)(bs / SS);
    const int    n  = idxbuf[m];
    float v;
    if (c < 3)            v = xyz[((size_t)b * NN + n) * 3 + c] - new_xyz[bs * 3 + c];
    else if (c < 3 + CIN) v = points[((size_t)b * CIN + (c - 3)) * NN + n];
    else                  v = 0.0f;
    A[tgl] = (__bf16)v;
  }
}

// ---------------- Weight f32[Cout,Cin] -> bf16[Cout,Cpad] (zero pad) -------
__global__ void wconv_kernel(const float* __restrict__ w,
                             __bf16* __restrict__ wbf,
                             int Cin, int Cpad, int total) {
  const int stride = gridDim.x * 256;
  for (int tgl = blockIdx.x * 256 + threadIdx.x; tgl < total; tgl += stride) {
    const int o = tgl / Cpad;
    const int c = tgl % Cpad;
    wbf[tgl] = (__bf16)((c < Cin) ? w[o * Cin + c] : 0.0f);
  }
}

// ---------------- bf16 WMMA GEMM -------------------------------------------
// Out[M,COUT] = A[M,CPAD] x W[COUT,CPAD]^T.
// Block = 8 waves; W staged in LDS (row stride CPAD+8 -> conflict-free
// ds_load_b128 B-fragments). Each wave owns 16 rows of A: all CPAD/32
// A-fragments are cached in registers, then reused across every n-tile,
// so A is fetched from HBM exactly once. 16x64 strip per wave (4 accs).
template <int CPAD, int COUT>
__global__ void wmma_gemm_kernel(const __bf16* __restrict__ A,
                                 const __bf16* __restrict__ W,
                                 __bf16* __restrict__ Out, int M) {
  constexpr int LSTR   = CPAD + 8;     // padded LDS row stride (keeps 16B align)
  constexpr int KSTEPS = CPAD / 32;
  constexpr int NCH    = COUT / 64;
  __shared__ __bf16 sW[COUT * LSTR];

  const int t = threadIdx.x;
  // cooperative W -> LDS (16-byte chunks)
  constexpr int CHUNKS = COUT * CPAD / 8;
  for (int q = t; q < CHUNKS; q += 256) {
    const int row = q / (CPAD / 8);
    const int c8  = q % (CPAD / 8);
    *(v8bf*)(sW + row * LSTR + c8 * 8) = *(const v8bf*)(W + row * CPAD + c8 * 8);
  }
  __syncthreads();

  const int lane  = t & 31;
  const int wave  = t >> 5;
  const int g     = lane >> 4;         // half-wave group
  const int mr    = lane & 15;         // row (A) / col (B) within tile
  const int mBase = (blockIdx.x * 8 + wave) * 16;
  if (mBase >= M) return;              // wave-uniform; after syncthreads

  // cache all A fragments for this wave's 16 rows
  v16bf aF[KSTEPS];
  const __bf16* arow = A + (size_t)(mBase + mr) * CPAD;
#pragma unroll
  for (int ks = 0; ks < KSTEPS; ++ks) {
    const v8bf lo = *(const v8bf*)(arow + ks * 32 + 8 * g);
    const v8bf hi = *(const v8bf*)(arow + ks * 32 + 16 + 8 * g);
#pragma unroll
    for (int i = 0; i < 8; ++i) { aF[ks][i] = lo[i]; aF[ks][i + 8] = hi[i]; }
  }

#pragma unroll
  for (int nc = 0; nc < NCH; ++nc) {
    v8f acc[4] = {{}, {}, {}, {}};
#pragma unroll
    for (int ks = 0; ks < KSTEPS; ++ks) {
#pragma unroll
      for (int nt = 0; nt < 4; ++nt) {
        const int nRow = nc * 64 + nt * 16 + mr;
        const __bf16* bp = sW + nRow * LSTR + ks * 32 + 8 * g;
        const v8bf lo = *(const v8bf*)(bp);
        const v8bf hi = *(const v8bf*)(bp + 16);
        v16bf bF;
#pragma unroll
        for (int i = 0; i < 8; ++i) { bF[i] = lo[i]; bF[i + 8] = hi[i]; }
        acc[nt] = __builtin_amdgcn_wmma_f32_16x16x32_bf16(
            /*neg_a=*/false, aF[ks], /*neg_b=*/false, bF,
            /*c_mod=*/(short)0, acc[nt], /*reuse_a=*/false, /*reuse_b=*/false);
      }
    }
#pragma unroll
    for (int nt = 0; nt < 4; ++nt) {
#pragma unroll
      for (int r = 0; r < 8; ++r) {
        Out[(size_t)(mBase + r + 8 * g) * COUT + nc * 64 + nt * 16 + mr] =
            (__bf16)acc[nt][r];
      }
    }
  }
}

// ---------------- BN statistics, stage 1: coalesced per-block partials -----
// 256 blocks; block g reduces rows [g*M/256, (g+1)*M/256). Thread t owns
// channel t%Cout, row-subset t/Cout -> consecutive lanes read consecutive
// channels (coalesced). Deterministic (no atomics).
__global__ void stats_partial_kernel(const __bf16* __restrict__ raw,
                                     float* __restrict__ part,  // [2][256][Cout]
                                     int M, int Cout) {
  __shared__ float lsum[256];
  __shared__ float lsq[256];
  const int t = threadIdx.x;
  const int gblk = blockIdx.x;          // 0..255
  const int rowsPer = M / 256;
  const int rep  = 256 / Cout;          // 4, 2 or 1
  const int c    = t % Cout;
  const int rsub = t / Cout;
  float s = 0.0f, q = 0.0f;
  const size_t base = (size_t)gblk * rowsPer * Cout;
  for (int r = rsub; r < rowsPer; r += rep) {
    const float x = (float)raw[base + (size_t)r * Cout + c];
    s += x; q += x * x;
  }
  lsum[t] = s; lsq[t] = q;
  __syncthreads();
  for (int off = 128; off >= Cout; off >>= 1) {
    if (t < off) { lsum[t] += lsum[t + off]; lsq[t] += lsq[t + off]; }
    __syncthreads();
  }
  if (t < Cout) {
    part[(size_t)gblk * Cout + t]         = lsum[t];
    part[(size_t)(256 + gblk) * Cout + t] = lsq[t];
  }
}

// ---------------- BN statistics, stage 2: one block per channel ------------
__global__ void stats_final_kernel(const float* __restrict__ part,
                                   float* __restrict__ sums,
                                   float* __restrict__ sumsq, int Cout) {
  __shared__ float sv[256];
  __shared__ float sq2[256];
  const int c = blockIdx.x;
  const int t = threadIdx.x;
  sv[t]  = part[(size_t)t * Cout + c];
  sq2[t] = part[(size_t)(256 + t) * Cout + c];
  __syncthreads();
  for (int off = 128; off > 0; off >>= 1) {
    if (t < off) { sv[t] += sv[t + off]; sq2[t] += sq2[t + off]; }
    __syncthreads();
  }
  if (t == 0) { sums[c] = sv[0]; sumsq[c] = sq2[0]; }
}

// ---------------- BN(train) + ReLU -> bf16 activations for next layer ------
__global__ void bn_relu_kernel(const __bf16* __restrict__ raw,
                               __bf16* __restrict__ outA,
                               const float* __restrict__ sums,
                               const float* __restrict__ sumsq,
                               const float* __restrict__ gamma,
                               const float* __restrict__ beta,
                               int M, int Cout) {
  const size_t total  = (size_t)M * Cout;
  const size_t stride = (size_t)gridDim.x * 256u;
  const float  invM   = 1.0f / (float)M;
  for (size_t tgl = (size_t)blockIdx.x * 256u + threadIdx.x; tgl < total; tgl += stride) {
    const int   c   = (int)(tgl % Cout);
    const float mu  = sums[c] * invM;
    const float var = sumsq[c] * invM - mu * mu;
    const float x   = (float)raw[tgl];
    const float y   = (x - mu) * rsqrtf(var + BN_EPS) * gamma[c] + beta[c];
    outA[tgl] = (__bf16)fmaxf(y, 0.0f);
  }
}

// ---------------- Max over K neighbors -> d_out new_points region ----------
__global__ void maxpool_kernel(const __bf16* __restrict__ act,
                               float* __restrict__ d_out,
                               int K, int Cout, int chOff) {
  const int total  = BB * SS * Cout;
  const int stride = gridDim.x * 256;
  for (int tgl = blockIdx.x * 256 + threadIdx.x; tgl < total; tgl += stride) {
    const int c  = tgl % Cout;
    const int bs = tgl / Cout;          // b*SS + s
    const int b  = bs / SS;
    const int s  = bs % SS;
    const __bf16* base = act + ((size_t)bs * K) * Cout + c;
    float v = -3.0e38f;
    for (int k = 0; k < K; ++k) v = fmaxf(v, (float)base[(size_t)k * Cout]);
    d_out[BB * SS * 3 + ((size_t)b * 640 + chOff + c) * SS + s] = v;
  }
}

// ---------------- new_xyz -> d_out[0 .. B*S*3) ------------------------------
__global__ void copy_xyz_kernel(const float* __restrict__ nxyz,
                                float* __restrict__ d_out) {
  const int t = blockIdx.x * 256 + threadIdx.x;
  if (t < BB * SS * 3) d_out[t] = nxyz[t];
}

// ---------------------------------------------------------------------------
static void launch_gemm(const __bf16* A, const __bf16* W, __bf16* Out,
                        int M, int Cpad, int Cout, hipStream_t stream) {
  const int blocks = M / 128;           // M is a multiple of 128 for all scales
  if      (Cpad ==  96 && Cout ==  64) wmma_gemm_kernel< 96,  64><<<blocks, 256, 0, stream>>>(A, W, Out, M);
  else if (Cpad ==  64 && Cout ==  64) wmma_gemm_kernel< 64,  64><<<blocks, 256, 0, stream>>>(A, W, Out, M);
  else if (Cpad ==  64 && Cout == 128) wmma_gemm_kernel< 64, 128><<<blocks, 256, 0, stream>>>(A, W, Out, M);
  else if (Cpad ==  96 && Cout == 128) wmma_gemm_kernel< 96, 128><<<blocks, 256, 0, stream>>>(A, W, Out, M);
  else if (Cpad == 128 && Cout == 128) wmma_gemm_kernel<128, 128><<<blocks, 256, 0, stream>>>(A, W, Out, M);
  else if (Cpad == 128 && Cout == 256) wmma_gemm_kernel<128, 256><<<blocks, 256, 0, stream>>>(A, W, Out, M);
}

extern "C" void kernel_launch(void* const* d_in, const int* in_sizes, int n_in,
                              void* d_out, int out_size, void* d_ws, size_t ws_size,
                              hipStream_t stream) {
  (void)in_sizes; (void)n_in; (void)out_size; (void)ws_size;
  const float* xyz    = (const float*)d_in[0];
  const float* points = (const float*)d_in[1];
  // params flattened: d_in[2 + (scale*3 + layer)*3 + {w,gamma,beta}]

  static const int mlp[3][3]   = {{64, 64, 128}, {128, 128, 256}, {128, 128, 256}};
  static const int nsamp[3]    = {16, 32, 128};
  static const float radii2[3] = {0.01f, 0.04f, 0.16f};
  static const int choff[3]    = {0, 128, 384};

  // ---- workspace carve-out (~540 MB peak) ----
  char* ws = (char*)d_ws;
  size_t off = 0;
  auto take = [&](size_t bytes) -> void* {
    void* p = ws + off;
    off += (bytes + 255) & ~(size_t)255;
    return p;
  };
  int*    fps_idx = (int*)take((size_t)BB * SS * sizeof(int));
  float*  nxyz    = (float*)take((size_t)BB * SS * 3 * sizeof(float));
  int*    idxbuf  = (int*)take((size_t)BB * SS * 128 * sizeof(int));
  float*  sums    = (float*)take(256 * sizeof(float));
  float*  sumsq   = (float*)take(256 * sizeof(float));
  float*  part    = (float*)take((size_t)2 * 256 * 256 * sizeof(float));
  __bf16* wbf[3][3];
  for (int s = 0; s < 3; ++s)
    for (int l = 0; l < 3; ++l) {
      const int cpad = (l == 0) ? 96 : mlp[s][l - 1];
      wbf[s][l] = (__bf16*)take((size_t)mlp[s][l] * cpad * sizeof(__bf16));
    }
  const size_t bufElems = (size_t)BB * SS * 128 * 256;  // max M * max Cout
  __bf16* buf0 = (__bf16*)take(bufElems * sizeof(__bf16));  // activations (GEMM A)
  __bf16* buf1 = (__bf16*)take(bufElems * sizeof(__bf16));  // raw GEMM output

  // ---- FPS + centroid coords ----
  fps_kernel<<<BB, 256, 0, stream>>>(xyz, fps_idx, nxyz);
  copy_xyz_kernel<<<(BB * SS * 3 + 255) / 256, 256, 0, stream>>>(nxyz, (float*)d_out);

  // ---- weights -> bf16 (padded) ----
  for (int s = 0; s < 3; ++s)
    for (int l = 0; l < 3; ++l) {
      const int cin  = (l == 0) ? 67 : mlp[s][l - 1];
      const int cpad = (l == 0) ? 96 : mlp[s][l - 1];
      const int cout = mlp[s][l];
      const float* wp = (const float*)d_in[2 + (s * 3 + l) * 3 + 0];
      const int total = cout * cpad;
      wconv_kernel<<<(total + 255) / 256, 256, 0, stream>>>(wp, wbf[s][l], cin, cpad, total);
    }

  // ---- per-scale pipeline ----
  for (int s = 0; s < 3; ++s) {
    const int K = nsamp[s];
    const int M = BB * SS * K;

    ball_query_kernel<<<BB * SS, 256, 0, stream>>>(xyz, nxyz, idxbuf, radii2[s], K);
    gather_kernel<<<4096, 256, 0, stream>>>(xyz, points, nxyz, idxbuf, buf0,
                                            K, 96, (size_t)M * 96);

    for (int l = 0; l < 3; ++l) {
      const int Cpad = (l == 0) ? 96 : mlp[s][l - 1];
      const int Cout = mlp[s][l];
      const float* gp = (const float*)d_in[2 + (s * 3 + l) * 3 + 1];
      const float* bp = (const float*)d_in[2 + (s * 3 + l) * 3 + 2];

      launch_gemm(buf0, wbf[s][l], buf1, M, Cpad, Cout, stream);
      stats_partial_kernel<<<256, 256, 0, stream>>>(buf1, part, M, Cout);
      stats_final_kernel<<<Cout, 256, 0, stream>>>(part, sums, sumsq, Cout);
      bn_relu_kernel<<<4096, 256, 0, stream>>>(buf1, buf0, sums, sumsq, gp, bp, M, Cout);
    }
    maxpool_kernel<<<(BB * SS * mlp[s][2] + 255) / 256, 256, 0, stream>>>(
        buf0, (float*)d_out, K, mlp[s][2], choff[s]);
  }
}
